// marker_re_model_81166291960414
// MI455X (gfx1250) — matrix operand order repaired and checked by reading, not compiled, not run on hardware
//
#include <hip/hip_runtime.h>
#include <hip/hip_bf16.h>
#include <math.h>

typedef __attribute__((ext_vector_type(16))) __bf16 v16bf;
typedef __attribute__((ext_vector_type(8)))  __bf16 v8bf;
typedef __attribute__((ext_vector_type(4)))  __bf16 v4bf;
typedef __attribute__((ext_vector_type(8)))  float  v8f;

#define B_    16
#define S_    2048
#define H_    768
#define MAXN  400
#define LAB   97
#define MID_  768
#define ROWS  (B_*MAXN)   /* 6400 */
#define K1    (2*H_)      /* 1536 */
#define NPAD  112         /* LAB padded to multiple of 16 */

#define MT    2           /* M-tiles per wave in GEMM1 */
#define NTL   4           /* N-tiles per block strip (64 cols) in GEMM1 */

/* low 32 bits of a generic pointer to __shared__ == LDS byte offset (ISA 10.2) */
__device__ __forceinline__ unsigned lds_off(const void* p) {
    return (unsigned)(unsigned long long)p;
}

/* ---------------- prep: zero accumulators ---------------- */
__global__ void init_acc(float* acc) {
    if (threadIdx.x < 2) acc[threadIdx.x] = 0.0f;
}

/* ---------------- prep: W1 (1536x768) -> W1T bf16 (768x1536) ---------------- */
__global__ void conv_w1t(const float* __restrict__ W1, __bf16* __restrict__ W1T) {
    const int total = K1 * MID_;
    for (int idx = blockIdx.x * blockDim.x + threadIdx.x; idx < total;
         idx += gridDim.x * blockDim.x) {
        const int n = idx / K1;       // 0..767
        const int k = idx % K1;       // 0..1535
        W1T[idx] = (__bf16)W1[(size_t)k * MID_ + n];
    }
}

/* ---------------- prep: W2 (768x97) -> W2T bf16 (112x768), zero padded ------ */
__global__ void conv_w2t(const float* __restrict__ W2, __bf16* __restrict__ W2T) {
    const int total = NPAD * MID_;
    for (int idx = blockIdx.x * blockDim.x + threadIdx.x; idx < total;
         idx += gridDim.x * blockDim.x) {
        const int n = idx / MID_;     // 0..111
        const int k = idx % MID_;     // 0..767
        W2T[idx] = (n < LAB) ? (__bf16)W2[(size_t)k * LAB + n] : (__bf16)0.0f;
    }
}

/* ---------------- gather marker vectors -> M bf16 (6400 x 1536) ------------- */
__global__ __launch_bounds__(192)
void gather_markers(const float* __restrict__ outs,
                    const int* __restrict__ rels_pos,
                    __bf16* __restrict__ M) {
    const int p = blockIdx.x;                 // 0..6399
    const int b = p / MAXN, i = p % MAXN;
    const int start = rels_pos[2 * b];
    int pos1 = start + 4 * i;
    int pos2 = pos1 + 2;
    pos1 = min(max(pos1, 0), S_ - 1);
    pos2 = min(max(pos2, 0), S_ - 1);
    const float* s1 = outs + ((size_t)b * S_ + pos1) * H_;
    const float* s2 = outs + ((size_t)b * S_ + pos2) * H_;
    __bf16* dst = M + (size_t)p * K1;
    const int e = threadIdx.x * 4;            // 192 threads * 4 = 768
    const float4 f1 = *(const float4*)(s1 + e);
    const float4 f2 = *(const float4*)(s2 + e);
    v4bf o1 = {(__bf16)f1.x, (__bf16)f1.y, (__bf16)f1.z, (__bf16)f1.w};
    v4bf o2 = {(__bf16)f2.x, (__bf16)f2.y, (__bf16)f2.z, (__bf16)f2.w};
    *(v4bf*)(dst + e)      = o1;
    *(v4bf*)(dst + H_ + e) = o2;
}

/* ================= GEMM1 with async B-staging through LDS ===================
 * C(6400x768) = relu(M(6400x1536) @ W1 + b1), bf16 out.
 * Block = 4 waves, all on one 64-col N-strip; wave owns 2 M-tiles (32x64).
 * B strip (64 cols x 32 K = 4KB bf16) is staged per block into LDS with
 * global_load_async_to_lds_b128 (ASYNCcnt), double-buffered so the fill of
 * buffer k+1 overlaps the 8 WMMAs of buffer k. LDS layout is fragment-linear:
 * thread (t*32+lane) stages exactly the 32B fragment lane needs for tile t. */
__global__ __launch_bounds__(128)
void wmma_gemm1_lds(const __bf16* __restrict__ A, const __bf16* __restrict__ BT,
                    const float* __restrict__ bias, __bf16* __restrict__ Out) {
    __shared__ __attribute__((aligned(32))) __bf16 ldsB[2][NTL * 32 * 16]; /* 2x4KB */

    const int tid  = threadIdx.x;
    const int wv   = tid >> 5;
    const int lane = tid & 31;
    const int half = lane >> 4;               /* K-half selector */
    const int nsub = lane & 15;

    const int nstrips = MID_ / (16 * NTL);    /* 12 */
    const int mg = blockIdx.x / nstrips;      /* 0..49, 8 mtiles per group */
    const int ns = blockIdx.x % nstrips;

    /* staging source/dest for this thread: tile t = wv, lane = lane */
    const int stCol = (ns * NTL + wv) * 16 + nsub;
    const __bf16* stSrc = BT + (size_t)stCol * K1 + 16 * half;
    const unsigned stDst0 = lds_off(&ldsB[0][tid * 16]);
    const unsigned stDst1 = lds_off(&ldsB[1][tid * 16]);

#define STAGE_B(k0, dst)                                                        \
    do {                                                                        \
        unsigned long long g_ = (unsigned long long)(stSrc + (k0));             \
        asm volatile("global_load_async_to_lds_b128 %0, %1, off"                \
                     :: "v"(dst), "v"(g_) : "memory");                          \
        asm volatile("global_load_async_to_lds_b128 %0, %1, off offset:16"      \
                     :: "v"(dst), "v"(g_) : "memory");                          \
    } while (0)
#define WAIT_ASYNC() asm volatile("s_wait_asynccnt 0x0" ::: "memory")

    /* A row pointers: wave wv owns mtiles mg*8 + wv*MT + m */
    const __bf16* Arow[MT];
#pragma unroll
    for (int m = 0; m < MT; ++m)
        Arow[m] = A + (size_t)((mg * (4 * MT) + wv * MT + m) * 16 + nsub) * K1;

    const v8f vzero = {0.f, 0.f, 0.f, 0.f, 0.f, 0.f, 0.f, 0.f};
    v8f acc[MT][NTL];
#pragma unroll
    for (int m = 0; m < MT; ++m)
#pragma unroll
        for (int t = 0; t < NTL; ++t) acc[m][t] = vzero;

    STAGE_B(0, stDst0);
    WAIT_ASYNC();
    __syncthreads();

    for (int k0 = 0; k0 < K1; k0 += 32) {
        const int buf = (k0 >> 5) & 1;
        if (k0 + 32 < K1) STAGE_B(k0 + 32, buf ? stDst0 : stDst1);  /* overlap */

        v16bf af[MT];
#pragma unroll
        for (int m = 0; m < MT; ++m) {
            v8bf a0 = *reinterpret_cast<const v8bf*>(Arow[m] + k0 + half * 16);
            v8bf a1 = *reinterpret_cast<const v8bf*>(Arow[m] + k0 + half * 16 + 8);
#pragma unroll
            for (int j = 0; j < 8; ++j) { af[m][j] = a0[j]; af[m][j + 8] = a1[j]; }
        }
#pragma unroll
        for (int t = 0; t < NTL; ++t) {
            const v16bf bfrag =
                *reinterpret_cast<const v16bf*>(&ldsB[buf][(t * 32 + lane) * 16]);
#pragma unroll
            for (int m = 0; m < MT; ++m)
                acc[m][t] = __builtin_amdgcn_wmma_f32_16x16x32_bf16(
                    false, af[m], false, bfrag, (short)0, acc[m][t], false, false);
        }
        WAIT_ASYNC();       /* staging of next buffer complete (this wave)   */
        __syncthreads();    /* ... and visible to all waves                  */
    }

#pragma unroll
    for (int t = 0; t < NTL; ++t) {
        const int ncolG = (ns * NTL + t) * 16 + nsub;
        const float bv = bias[ncolG];
#pragma unroll
        for (int m = 0; m < MT; ++m) {
            const int rowBase = (mg * (4 * MT) + wv * MT + m) * 16;
#pragma unroll
            for (int r = 0; r < 8; ++r) {
                float v = acc[m][t][r] + bv;
                v = v > 0.f ? v : 0.f;
                Out[(size_t)(rowBase + r + 8 * half) * MID_ + ncolG] = (__bf16)v;
            }
        }
    }
#undef STAGE_B
#undef WAIT_ASYNC
}

/* ---------------- generic WMMA GEMM (used for GEMM2, f32 logits out) -------- */
template<int NT>
__global__ __launch_bounds__(128)
void wmma_gemm_f32(const __bf16* __restrict__ A, const __bf16* __restrict__ BT,
                   const float* __restrict__ bias, int nReal,
                   float* __restrict__ Of32, int Mrows, int K, int N) {
    const int wave    = blockIdx.x * (blockDim.x >> 5) + (threadIdx.x >> 5);
    const int lane    = threadIdx.x & 31;
    const int nstrips = N / (16 * NT);
    const int mtiles  = Mrows >> 4;
    if (wave >= mtiles * nstrips) return;     // wave-uniform
    const int mtile  = wave / nstrips;
    const int nstrip = wave % nstrips;
    const int half   = lane >> 4;
    const int nsub   = lane & 15;

    const __bf16* Arow = A + (size_t)(mtile * 16 + nsub) * K;

    const v8f vzero = {0.f, 0.f, 0.f, 0.f, 0.f, 0.f, 0.f, 0.f};
    v8f acc[NT];
#pragma unroll
    for (int t = 0; t < NT; ++t) acc[t] = vzero;

    for (int k0 = 0; k0 < K; k0 += 32) {
        v8bf a0 = *reinterpret_cast<const v8bf*>(Arow + k0 + half * 16);
        v8bf a1 = *reinterpret_cast<const v8bf*>(Arow + k0 + half * 16 + 8);
        v16bf afrag;
#pragma unroll
        for (int j = 0; j < 8; ++j) { afrag[j] = a0[j]; afrag[j + 8] = a1[j]; }
#pragma unroll
        for (int t = 0; t < NT; ++t) {
            const int ncolG = (nstrip * NT + t) * 16 + nsub;
            v16bf bfrag = *reinterpret_cast<const v16bf*>(
                BT + (size_t)ncolG * K + k0 + half * 16);
            acc[t] = __builtin_amdgcn_wmma_f32_16x16x32_bf16(
                false, afrag, false, bfrag, (short)0, acc[t], false, false);
        }
    }
#pragma unroll
    for (int t = 0; t < NT; ++t) {
        const int ncolG = (nstrip * NT + t) * 16 + nsub;
        const float bv = (ncolG < nReal) ? bias[ncolG] : 0.0f;
#pragma unroll
        for (int r = 0; r < 8; ++r) {
            const int rowG = mtile * 16 + r + 8 * half;
            Of32[(size_t)rowG * N + ncolG] = acc[t][r] + bv;
        }
    }
}

/* ---------------- per-row log-softmax / NLL / argmax (wave per row) --------- */
__global__ __launch_bounds__(128)
void softmax_nll(const float* __restrict__ logits, const int* __restrict__ rels,
                 const int* __restrict__ rels_pos, float* __restrict__ outPred,
                 float* __restrict__ outLab, float* __restrict__ acc) {
    const int row  = blockIdx.x * (blockDim.x >> 5) + (threadIdx.x >> 5);
    const int lane = threadIdx.x & 31;
    if (row >= ROWS) return;
    const int b = row / MAXN, n = row % MAXN;
    const int num   = rels_pos[2 * b + 1];
    const bool mask = n < num;
    const int label = rels[b * MAXN + n];
    const float* lrow = logits + (size_t)row * NPAD;

    float v[4];
    float bestv = -INFINITY; int besti = 0;
#pragma unroll
    for (int i = 0; i < 4; ++i) {
        const int c = lane + 32 * i;
        v[i] = (c < LAB) ? lrow[c] : -INFINITY;
        if (v[i] > bestv) { bestv = v[i]; besti = c; }
    }
#pragma unroll
    for (int off = 16; off > 0; off >>= 1) {   // wave32 max + first-argmax
        const float ov = __shfl_xor(bestv, off);
        const int   oi = __shfl_xor(besti, off);
        if (ov > bestv || (ov == bestv && oi < besti)) { bestv = ov; besti = oi; }
    }
    const float maxv = bestv;
    float sum = 0.f, lv = 0.f;
#pragma unroll
    for (int i = 0; i < 4; ++i) {
        const int c = lane + 32 * i;
        if (c < LAB) {
            sum += expf(v[i] - maxv);
            if (c == label) lv = v[i];
        }
    }
#pragma unroll
    for (int off = 16; off > 0; off >>= 1) {
        sum += __shfl_xor(sum, off);
        lv  += __shfl_xor(lv, off);
    }
    if (lane == 0) {
        outPred[row] = mask ? (float)besti : -1.0f;
        outLab[row]  = mask ? (float)label : -1.0f;
        if (mask) {
            const float nll = (maxv + logf(sum)) - lv;
            atomicAdd(acc + 0, nll);
            atomicAdd(acc + 1, 1.0f);
        }
    }
}

__global__ void finalize(const float* acc, float* outLoss) {
    outLoss[0] = acc[0] / acc[1];
}

/* ---------------- launch ---------------- */
extern "C" void kernel_launch(void* const* d_in, const int* in_sizes, int n_in,
                              void* d_out, int out_size, void* d_ws, size_t ws_size,
                              hipStream_t stream) {
    const float* outs     = (const float*)d_in[0];
    const int*   rels     = (const int*)d_in[1];
    const int*   rels_pos = (const int*)d_in[2];
    const float* W1       = (const float*)d_in[3];
    const float* b1       = (const float*)d_in[4];
    const float* W2       = (const float*)d_in[5];
    const float* b2       = (const float*)d_in[6];

    char* ws = (char*)d_ws;                       /* 256B-aligned sub-buffers */
    __bf16* M    = (__bf16*)(ws + 0);             /* 6400x1536 bf16 = 19660800 */
    __bf16* W1T  = (__bf16*)(ws + 19660800);      /* 768x1536  bf16 =  2359296 */
    __bf16* W2T  = (__bf16*)(ws + 22020096);      /* 112x768   bf16 =   172032 */
    __bf16* Hbuf = (__bf16*)(ws + 22192128);      /* 6400x768  bf16 =  9830400 */
    float*  L    = (float*)(ws + 32022528);       /* 6400x112  f32  =  2867200 */
    float*  acc  = (float*)(ws + 34889728);       /* 2 floats */

    float* out     = (float*)d_out;
    float* outLoss = out;                         /* [0] */
    float* outPred = out + 1;                     /* [1 .. 6400] */
    float* outLab  = out + 1 + ROWS;              /* [6401 .. 12800] */

    init_acc<<<1, 32, 0, stream>>>(acc);
    conv_w1t<<<1152, 256, 0, stream>>>(W1, W1T);
    conv_w2t<<<336, 256, 0, stream>>>(W2, W2T);
    gather_markers<<<ROWS, 192, 0, stream>>>(outs, rels_pos, M);

    /* GEMM1: 50 m-groups (8 mtiles each) x 12 n-strips = 600 blocks, 4 waves */
    wmma_gemm1_lds<<<600, 128, 0, stream>>>(M, W1T, b1, Hbuf);

    /* GEMM2: (6400x768)x(768x112) + b2 -> f32 logits; 400 waves, 100 blocks  */
    wmma_gemm_f32<7><<<100, 128, 0, stream>>>(Hbuf, W2T, b2, LAB, L,
                                              ROWS, MID_, NPAD);

    softmax_nll<<<ROWS / 4, 128, 0, stream>>>(L, rels, rels_pos, outPred, outLab, acc);
    finalize<<<1, 1, 0, stream>>>(acc, outLoss);
}
